// RNNInferenceNetwork_73134703116465
// MI455X (gfx1250) — matrix operand order repaired and checked by reading, not compiled, not run on hardware
//
#include <hip/hip_runtime.h>
#include <stdint.h>

// ---------------------------------------------------------------------------
// Persistent GRU variational-RNN for MI455X (gfx1250, wave32, WMMA bf16).
//   - weights packed once per launch to bf16, L2-resident (2.3 MB << 192 MB L2)
//   - 64 workgroups x 256 threads (8 waves); each WG owns 32 batch rows and
//     runs the full T=512 recurrence privately (batch rows are independent)
//   - all GEMMs via v_wmma_f32_16x16x32_bf16, f32 accumulate
//   - hidden state kept in registers (per-wave v8f accumulator layout)
//   - kt loops kept ROLLED (#pragma unroll 1) with loop-carried pointers:
//     avoids scratch spills (round 1) and per-iteration address recompute
//     (round 2); gate weights addressed via immediate-foldable offsets.
//   - y_{t+1}/noise prefetched a full step ahead; y staging folded into
//     Phase E (behind the B/C barriers) so no dedicated stall phase.
// ---------------------------------------------------------------------------

typedef __attribute__((ext_vector_type(16))) __bf16 v16bf;
typedef __attribute__((ext_vector_type(8)))  float  v8f;

#define T_STEPS 512
#define BATCH   256
#define OBS     128
#define HID     512
#define XDIM    64
#define INDIM   192          // OBS + XDIM
#define KTOT    704          // INDIM + HID
#define SBTOT   2048         // NUM_SAMPLES * BATCH
#define NGATE   1536         // 3*HID
#define ROWS    32           // batch rows per workgroup (2 WMMA M-tiles)
#define NWG     (SBTOT / ROWS)
#define NTHREADS 256
#define A_STRIDE 720         // bf16 elems per A row (704 + 16 pad, bank skew)
#define LOG_2PI 1.8378770664093453f

union ABFrag { v16bf v; uint4 q[2]; uint16_t u[16]; };

__device__ __forceinline__ uint16_t f2bf(float f) {
  union { float f; uint32_t u; } c; c.f = f;
  uint32_t u = c.u;
  u += 0x7FFFu + ((u >> 16) & 1u);          // round-to-nearest-even
  return (uint16_t)(u >> 16);
}

__device__ __forceinline__ float sigm(float x) { return 1.0f / (1.0f + __expf(-x)); }
__device__ __forceinline__ float tanh_fast(float x) {
  float e = __expf(-2.0f * x);
  return (1.0f - e) / (1.0f + e);
}
__device__ __forceinline__ v8f splat8(float x) { return (v8f){x, x, x, x, x, x, x, x}; }

// A fragment (16x32 bf16, ISA 7.12.2): per-lane chunks at +0 and +16 elems.
__device__ __forceinline__ v16bf load_a_p(const uint16_t* p) {
  ABFrag f;
  f.q[0] = *(const uint4*)(p);
  f.q[1] = *(const uint4*)(p + 16);
  return f.v;
}
// B fragment (32x16 bf16): per-lane chunks at +0 and +16 elems (matches A).
__device__ __forceinline__ v16bf load_b_p(const uint16_t* p) {
  ABFrag f;
  f.q[0] = *(const uint4*)(p);
  f.q[1] = *(const uint4*)(p + 16);
  return f.v;
}

#define WMMA_BF16(A_, B_, C_) \
  __builtin_amdgcn_wmma_f32_16x16x32_bf16(false, (A_), false, (B_), (short)0, (C_), false, false)

// ------------------------- weight packing kernels --------------------------

__global__ void pack_wcat(const float* __restrict__ Wih, const float* __restrict__ Whh,
                          uint16_t* __restrict__ out) {
  int i = blockIdx.x * blockDim.x + threadIdx.x;
  if (i >= NGATE * KTOT) return;
  int n = i / KTOT, k = i % KTOT;
  float v = (k < INDIM) ? Wih[(size_t)n * INDIM + k] : Whh[(size_t)n * HID + (k - INDIM)];
  out[i] = f2bf(v);
}

__global__ void pack_wls(const float* __restrict__ Wloc, const float* __restrict__ Wsc,
                         uint16_t* __restrict__ out) {
  int i = blockIdx.x * blockDim.x + threadIdx.x;
  if (i >= 2 * XDIM * HID) return;
  int nn = i / HID, k = i % HID;
  float v = (nn < XDIM) ? Wloc[(size_t)k * XDIM + nn] : Wsc[(size_t)k * XDIM + (nn - XDIM)];
  out[i] = f2bf(v);
}

__global__ void pack_bias(const float* __restrict__ bih, const float* __restrict__ bhh,
                          const float* __restrict__ bloc, const float* __restrict__ bsc,
                          float* __restrict__ bsum, float* __restrict__ bi,
                          float* __restrict__ bh, float* __restrict__ bls) {
  int i = blockIdx.x * blockDim.x + threadIdx.x;
  if (i < NGATE) { bsum[i] = bih[i] + bhh[i]; bi[i] = bih[i]; bh[i] = bhh[i]; }
  if (i < 2 * XDIM) bls[i] = (i < XDIM) ? bloc[i] : bsc[i - XDIM];
}

// --------------------------- persistent kernel -----------------------------

__global__ __launch_bounds__(NTHREADS, 1) void gru_persistent(
    const float* __restrict__ y, const float* __restrict__ noise,
    const uint16_t* __restrict__ Wcat, const uint16_t* __restrict__ Wls,
    const float* __restrict__ bsum, const float* __restrict__ bi,
    const float* __restrict__ bh, const float* __restrict__ bls,
    float* __restrict__ out)
{
  __shared__ uint16_t Abuf[ROWS * A_STRIDE];     // [y_t | l_{t-1} | h_{t-1}] bf16
  __shared__ float lsbuf[ROWS * 2 * XDIM];       // loc | scale-pre (f32)
  __shared__ float redbuf[ROWS * 8 * 2];         // row-reduction partials

  const int tid  = threadIdx.x;
  const int lane = tid & 31;
  const int wv   = tid >> 5;                     // wave id 0..7
  const int r0   = blockIdx.x * ROWS;            // first SB row of this WG
  const int hi   = (lane & 16) ? 8 : 0;          // D-frag row offset
  const int col  = lane & 15;                    // D-frag column within tile

  // zero l/h regions (and y, harmlessly)
  for (int i = tid; i < ROWS * A_STRIDE; i += NTHREADS) Abuf[i] = 0;

  // persistent hidden state in registers: hold[j][m] covers
  // rows m*16..m*16+15, hidden cols (wv+8j)*16..+16 in D-fragment layout
  v8f hold[4][2];
#pragma unroll
  for (int j = 0; j < 4; ++j)
#pragma unroll
    for (int m = 0; m < 2; ++m) hold[j][m] = splat8(0.0f);

  float biasr[4], biasz[4], biasin[4], biashn[4];
#pragma unroll
  for (int j = 0; j < 4; ++j) {
    const int c = (wv + 8 * j) * 16 + col;
    biasr[j]  = bsum[c];                // r gate:   i_r + h_r fused
    biasz[j]  = bsum[HID + c];          // z gate:   i_z + h_z fused
    biasin[j] = bi[2 * HID + c];        // n gate, input part
    biashn[j] = bh[2 * HID + c];        // n gate, hidden part (scaled by r)
  }
  const float bias_ls = bls[wv * 16 + col];

  // per-lane base pointers for the GEMM fragments (loop-carried in kt loops)
  const uint16_t* const apBase =
      Abuf + (lane & 15) * A_STRIDE + ((lane & 16) ? 8 : 0);
  const int boff = (lane & 16) ? 8 : 0;           // B-frag K sub-offset

  // y staging map: thread -> (row, 16 obs cols)
  const int yrow = tid >> 3;
  const int yc0  = (tid & 7) * 16;
  const float* ybase = y + (size_t)((r0 + yrow) & (BATCH - 1)) * T_STEPS * OBS + yc0;
  uint16_t* const ydst = Abuf + yrow * A_STRIDE + yc0;

  // sample-phase map: thread -> (row, 8 xdim cols)
  const int srow = tid >> 3;
  const int sc0  = (tid & 7) * 8;
  const int ssb  = r0 + srow;

  const size_t XOUT = (size_t)SBTOT * T_STEPS * XDIM;
  const size_t EOUT = (size_t)SBTOT * T_STEPS;

  // ---- prologue: stage y_0 ----
  {
    const float4* yp = (const float4*)(ybase);
    ABFrag pk;
#pragma unroll
    for (int q = 0; q < 4; ++q) {
      float4 v = yp[q];
      pk.u[q * 4 + 0] = f2bf(v.x); pk.u[q * 4 + 1] = f2bf(v.y);
      pk.u[q * 4 + 2] = f2bf(v.z); pk.u[q * 4 + 3] = f2bf(v.w);
    }
    *(uint4*)(ydst)     = pk.q[0];
    *(uint4*)(ydst + 8) = pk.q[1];
  }
  __syncthreads();

#pragma unroll 1
  for (int t = 0; t < T_STEPS; ++t) {
    // one-step-ahead prefetch: y_{t+1} and this step's noise row
    if (t + 1 < T_STEPS)
      __builtin_prefetch(ybase + (size_t)(t + 1) * OBS, 0, 1);
    __builtin_prefetch(noise + ((size_t)t * SBTOT + ssb) * XDIM + sc0, 0, 1);

    // ---- Phase B: fused gate GEMMs (K=704) + GRU gating on accumulators ----
#pragma unroll
    for (int j = 0; j < 4; ++j) {
      const int jt = wv + 8 * j;
      v8f ar[2], az[2], ain[2], ahn[2];
#pragma unroll
      for (int m = 0; m < 2; ++m) {
        ar[m]  = splat8(biasr[j]);
        az[m]  = splat8(biasz[j]);
        ain[m] = splat8(biasin[j]);
        ahn[m] = splat8(biashn[j]);
      }
      const uint16_t* ap = apBase;
      const uint16_t* bp = Wcat + (size_t)(jt * 16 + col) * KTOT + boff;
      // K-tiles 0..5: rnn_in = [y | l]  (n-gate input accumulator)
#pragma unroll 1
      for (int kt = 0; kt < 6; ++kt) {
        const v16bf a0 = load_a_p(ap);
        const v16bf a1 = load_a_p(ap + 16 * A_STRIDE);
        const v16bf br = load_b_p(bp);
        const v16bf bz = load_b_p(bp + (size_t)HID * KTOT);
        const v16bf bn = load_b_p(bp + (size_t)2 * HID * KTOT);
        ar[0]  = WMMA_BF16(a0, br, ar[0]);   ar[1]  = WMMA_BF16(a1, br, ar[1]);
        az[0]  = WMMA_BF16(a0, bz, az[0]);   az[1]  = WMMA_BF16(a1, bz, az[1]);
        ain[0] = WMMA_BF16(a0, bn, ain[0]);  ain[1] = WMMA_BF16(a1, bn, ain[1]);
        ap += 32; bp += 32;
      }
      // K-tiles 6..21: h_{t-1}  (n-gate hidden accumulator)
#pragma unroll 1
      for (int kt = 6; kt < 22; ++kt) {
        const v16bf a0 = load_a_p(ap);
        const v16bf a1 = load_a_p(ap + 16 * A_STRIDE);
        const v16bf br = load_b_p(bp);
        const v16bf bz = load_b_p(bp + (size_t)HID * KTOT);
        const v16bf bn = load_b_p(bp + (size_t)2 * HID * KTOT);
        ar[0]  = WMMA_BF16(a0, br, ar[0]);   ar[1]  = WMMA_BF16(a1, br, ar[1]);
        az[0]  = WMMA_BF16(a0, bz, az[0]);   az[1]  = WMMA_BF16(a1, bz, az[1]);
        ahn[0] = WMMA_BF16(a0, bn, ahn[0]);  ahn[1] = WMMA_BF16(a1, bn, ahn[1]);
        ap += 32; bp += 32;
      }
      // GRU update, elementwise on accumulator layout (r/z/n tiles share it)
#pragma unroll
      for (int m = 0; m < 2; ++m)
#pragma unroll
        for (int e = 0; e < 8; ++e) {
          const float r = sigm(ar[m][e]);
          const float z = sigm(az[m][e]);
          const float n = tanh_fast(ain[m][e] + r * ahn[m][e]);
          hold[j][m][e] = (1.0f - z) * n + z * hold[j][m][e];
        }
    }
    __syncthreads();   // everyone done reading y_t / l_{t-1} / h_{t-1}

    // ---- Phase C: commit new hidden state (bf16) into Abuf[:, INDIM:KTOT) ----
#pragma unroll
    for (int j = 0; j < 4; ++j) {
      const int ccol = INDIM + (wv + 8 * j) * 16 + col;
#pragma unroll
      for (int m = 0; m < 2; ++m)
#pragma unroll
        for (int e = 0; e < 8; ++e)
          Abuf[(m * 16 + hi + e) * A_STRIDE + ccol] = f2bf(hold[j][m][e]);
    }
    __syncthreads();

    // ---- Phase D: loc/scale head GEMM  h(32x512) @ Wls^T(128x512) ----
    {
      v8f c0 = splat8(bias_ls), c1 = splat8(bias_ls);
      const uint16_t* ap = apBase + 6 * 32;
      const uint16_t* bp = Wls + (size_t)(wv * 16 + col) * HID + boff;
#pragma unroll 1
      for (int kt = 0; kt < 16; ++kt) {
        const v16bf a0 = load_a_p(ap);
        const v16bf a1 = load_a_p(ap + 16 * A_STRIDE);
        const v16bf b  = load_b_p(bp);
        c0 = WMMA_BF16(a0, b, c0);
        c1 = WMMA_BF16(a1, b, c1);
        ap += 32; bp += 32;
      }
#pragma unroll
      for (int e = 0; e < 8; ++e) {
        lsbuf[(hi + e) * 2 * XDIM + wv * 16 + col]      = c0[e];
        lsbuf[(16 + hi + e) * 2 * XDIM + wv * 16 + col] = c1[e];
      }
    }
    __syncthreads();

    // ---- Phase E: softplus, rsample, outputs, reductions, y_{t+1} staging ----
    {
      const float* ns = noise + ((size_t)t * SBTOT + ssb) * XDIM + sc0;
      float* xo = out + ((size_t)ssb * T_STEPS + t) * XDIM + sc0;
      float se = 0.0f, sl = 0.0f;
#pragma unroll
      for (int c = 0; c < 8; ++c) {
        const float loc = lsbuf[srow * 2 * XDIM + sc0 + c];
        const float pre = lsbuf[srow * 2 * XDIM + XDIM + sc0 + c];
        const float eps = ns[c];
        const float sc  = (pre > 15.0f) ? pre : log1pf(__expf(pre));   // softplus
        const float ll  = loc + sc * eps;                               // rsample
        Abuf[srow * A_STRIDE + OBS + sc0 + c] = f2bf(ll);               // next rnn_in
        xo[c] = ll;
        const float lg = __logf(sc);
        se += 0.5f * (1.0f + LOG_2PI) + lg;
        sl += -0.5f * eps * eps - lg - 0.5f * LOG_2PI;
      }
      redbuf[(srow * 8 + (tid & 7)) * 2 + 0] = se;
      redbuf[(srow * 8 + (tid & 7)) * 2 + 1] = sl;

      // stage y_{t+1} (prefetched a full step ago -> near-cache hit); the
      // y region of Abuf is free here: all Phase-B reads of y_t are behind
      // two barriers.
      if (t + 1 < T_STEPS) {
        const float4* yp = (const float4*)(ybase + (size_t)(t + 1) * OBS);
        ABFrag pk;
#pragma unroll
        for (int q = 0; q < 4; ++q) {
          float4 v = yp[q];
          pk.u[q * 4 + 0] = f2bf(v.x); pk.u[q * 4 + 1] = f2bf(v.y);
          pk.u[q * 4 + 2] = f2bf(v.z); pk.u[q * 4 + 3] = f2bf(v.w);
        }
        *(uint4*)(ydst)     = pk.q[0];
        *(uint4*)(ydst + 8) = pk.q[1];
      }
    }
    __syncthreads();
    if (tid < ROWS) {
      float se = 0.0f, sl = 0.0f;
#pragma unroll
      for (int p = 0; p < 8; ++p) {
        se += redbuf[(tid * 8 + p) * 2 + 0];
        sl += redbuf[(tid * 8 + p) * 2 + 1];
      }
      out[XOUT + (size_t)(r0 + tid) * T_STEPS + t]        = se;
      out[XOUT + EOUT + (size_t)(r0 + tid) * T_STEPS + t] = sl;
    }
    __syncthreads();
  }
}

// ------------------------------- launcher ----------------------------------

extern "C" void kernel_launch(void* const* d_in, const int* in_sizes, int n_in,
                              void* d_out, int out_size, void* d_ws, size_t ws_size,
                              hipStream_t stream) {
  const float* y     = (const float*)d_in[0];
  const float* W_ih  = (const float*)d_in[1];
  const float* W_hh  = (const float*)d_in[2];
  const float* b_ih  = (const float*)d_in[3];
  const float* b_hh  = (const float*)d_in[4];
  const float* W_loc = (const float*)d_in[5];
  const float* b_loc = (const float*)d_in[6];
  const float* W_sc  = (const float*)d_in[7];
  const float* b_sc  = (const float*)d_in[8];
  const float* noise = (const float*)d_in[9];

  // workspace layout (~2.3 MB)
  uint16_t* Wcat = (uint16_t*)d_ws;                 // [1536][704] bf16
  uint16_t* Wls  = Wcat + (size_t)NGATE * KTOT;     // [128][512] bf16
  float*    bsum = (float*)(Wls + (size_t)2 * XDIM * HID);
  float*    bi   = bsum + NGATE;
  float*    bh   = bi + NGATE;
  float*    bls  = bh + NGATE;

  pack_wcat<<<(NGATE * KTOT + 255) / 256, 256, 0, stream>>>(W_ih, W_hh, Wcat);
  pack_wls<<<(2 * XDIM * HID + 255) / 256, 256, 0, stream>>>(W_loc, W_sc, Wls);
  pack_bias<<<(NGATE + 255) / 256, 256, 0, stream>>>(b_ih, b_hh, b_loc, b_sc,
                                                     bsum, bi, bh, bls);
  gru_persistent<<<NWG, NTHREADS, 0, stream>>>(y, noise, Wcat, Wls,
                                               bsum, bi, bh, bls, (float*)d_out);
}
